// VQVAE_45165876084798
// MI455X (gfx1250) — matrix operand order repaired and checked by reading, not compiled, not run on hardware
//
#include <hip/hip_runtime.h>
#include <hip/hip_fp16.h>

// ---------------------------------------------------------------------------
// VQ-VAE forward for MI455X (gfx1250, wave32).
//  - All matmul-shaped stages use v_wmma_f32_16x16x32_f16 (f32 accumulate).
//  - Weights / codebook staged into LDS with the Tensor Data Mover.
//  - Boundary gathers use zero-page pointer-select: unconditional loads, no
//    exec-mask branches, no per-element waits.
// ---------------------------------------------------------------------------

typedef __attribute__((ext_vector_type(16))) _Float16 v16h;
typedef __attribute__((ext_vector_type(8)))  float    v8f;
typedef __attribute__((ext_vector_type(8)))  unsigned int v8u;
typedef __attribute__((ext_vector_type(4)))  unsigned int v4u;
typedef __attribute__((ext_vector_type(8)))  int      v8i;
typedef __attribute__((ext_vector_type(4)))  int      v4i;

// ---- LDS offset of a generic pointer (addrspacecast generic -> local) -----
__device__ inline unsigned lds_off_u32(const void* p) {
  const __attribute__((address_space(3))) char* lp =
      (const __attribute__((address_space(3))) char*)p;
  return (unsigned)(unsigned long long)lp;
}

// ---- Tensor Data Mover: global -> LDS, 1D (tile1==0) or 2D tile -----------
__device__ inline void tdm_load(void* ldsDst, const void* gsrc,
                                unsigned tile0_dw, unsigned tile1,
                                unsigned stride0_dw) {
  unsigned long long ga = (unsigned long long)gsrc;
  const unsigned td0 = stride0_dw;
  const unsigned td1 = tile1 ? tile1 : 1u;
  v4u g0;
  g0[0] = 1u;                                        // count=1, user descriptor
  g0[1] = lds_off_u32(ldsDst);                       // lds_addr (bytes)
  g0[2] = (unsigned)ga;                              // global_addr[31:0]
  g0[3] = (unsigned)((ga >> 32) & 0x1FFFFFFu) | (2u << 30);  // [56:32] + type=2
  v8i g1;
  g1[0] = (int)(2u << 16);                           // data_size = 4 bytes
  g1[1] = (int)((td0 & 0xFFFFu) << 16);              // tensor_dim0[15:0]
  g1[2] = (int)(((td0 >> 16) & 0xFFFFu) | ((td1 & 0xFFFFu) << 16));
  g1[3] = (int)(((td1 >> 16) & 0xFFFFu) | ((tile0_dw & 0xFFFFu) << 16));
  g1[4] = (int)(tile1 & 0xFFFFu);                    // tile_dim1 (0 => 1D)
  g1[5] = (int)stride0_dw;                           // tensor_dim0_stride lo
  g1[6] = 0;
  g1[7] = 0;
  v4i gz = {0, 0, 0, 0};
#if defined(__clang_major__) && __clang_major__ >= 23
  v8i gz8 = {0, 0, 0, 0, 0, 0, 0, 0};
  __builtin_amdgcn_tensor_load_to_lds(g0, g1, gz, gz, gz8, 0);
#else
  __builtin_amdgcn_tensor_load_to_lds(g0, g1, gz, gz, 0);
#endif
}

// ---- WMMA core ------------------------------------------------------------
__device__ inline v8f wmma_f16(v8f acc, v8u au, v8u bu) {
  return __builtin_amdgcn_wmma_f32_16x16x32_f16(
      false, __builtin_bit_cast(v16h, au),
      false, __builtin_bit_cast(v16h, bu), (short)0, acc, false, false);
}

__device__ inline unsigned packh2(_Float16 lo, _Float16 hi) {
  union { unsigned u; _Float16 h[2]; } r;
  r.h[0] = lo; r.h[1] = hi;
  return r.u;
}

__device__ inline v8u cat2(uint4 p, uint4 q) {
  v8u r;
  r[0] = p.x; r[1] = p.y; r[2] = p.z; r[3] = p.w;
  r[4] = q.x; r[5] = q.y; r[6] = q.z; r[7] = q.w;
  return r;
}

// A-row fetch: within a 32-wide k-window of a contiguous row, lanes 0-15 need
// dwords {0-3, 8-11}, lanes 16-31 need {4-7, 12-15}  ->  two 128-bit loads.
__device__ inline v8u fetch_a_row(const uint4* row4, int ksWin, int half) {
  return cat2(row4[ksWin * 4 + half], row4[ksWin * 4 + 2 + half]);
}
// B-column fetch (contiguous row holding this lane's column): lanes 0-15 need
// dwords {0-3, 8-11}, lanes 16-31 need {4-7, 12-15} -> two 128-bit loads.
__device__ inline v8u fetch_b_row(const uint4* row4, int ksWin, int half) {
  return cat2(row4[ksWin * 4 + half], row4[ksWin * 4 + 2 + half]);
}

// Scalar (gather) packers — element functor must be branch-free.
template <typename F>
__device__ inline v8u pack_a(int lane, int kOff, F&& f) {
  const int l15 = lane & 15, half = lane >> 4;
  v8u au;
#pragma unroll
  for (int j = 0; j < 8; ++j) {
    const int ka = kOff + (half ? 8 : 0) + ((j < 4) ? 2 * j : 16 + 2 * (j - 4));
    au[j] = packh2(f(l15, ka), f(l15, ka + 1));
  }
  return au;
}
template <typename F>
__device__ inline v8u pack_b(int lane, int kOff, F&& f) {
  const int l15 = lane & 15, half = lane >> 4;
  v8u bu;
#pragma unroll
  for (int j = 0; j < 8; ++j) {
    const int kb = kOff + (half ? 8 : 0) + ((j < 4) ? 2 * j : 16 + 2 * (j - 4));
    bu[j] = packh2(f(kb, l15), f(kb + 1, l15));
  }
  return bu;
}

// ---------------------------------------------------------------------------
// prep: fp32 weights/codebook -> f16, codebook squared norms, zero page
// ---------------------------------------------------------------------------
__global__ __launch_bounds__(256) void k_prep(const float* __restrict__ w1,
                                              const float* __restrict__ w2,
                                              const float* __restrict__ cb,
                                              const float* __restrict__ dw1,
                                              _Float16* __restrict__ w1h,
                                              _Float16* __restrict__ w2h,
                                              _Float16* __restrict__ cbh,
                                              _Float16* __restrict__ dw1h,
                                              float* __restrict__ cbn,
                                              float* __restrict__ zpg) {
  const int id = blockIdx.x * 256 + threadIdx.x;
  if (id < 3072) w1h[id] = (_Float16)w1[id];
  if (id < 65536) {
    w2h[id]  = (_Float16)w2[id];
    dw1h[id] = (_Float16)dw1[id];
  }
  if (id < 32768) cbh[id] = (_Float16)cb[id];
  if (id < 512) {
    float s = 0.f;
    for (int d = 0; d < 64; ++d) { const float e = cb[id * 64 + d]; s += e * e; }
    cbn[id] = s;
  }
  if (id < 64) zpg[id] = 0.f;    // 256-byte zero page for OOB gathers
}

// ---------------------------------------------------------------------------
// conv1: x(32,3,256,256) -> relu -> z1h f16 (32,64,128,128); K=48 pad to 64
// ---------------------------------------------------------------------------
__global__ __launch_bounds__(256) void k_conv1(const float* __restrict__ x,
                                               const _Float16* __restrict__ w1h,
                                               const float* __restrict__ b1,
                                               const float* __restrict__ zeroF,
                                               _Float16* __restrict__ z1h) {
  __shared__ __align__(16) _Float16 lw[64 * 48];   // 6 KB; 96B rows (16B mult)
  if (threadIdx.x < 32) {
    tdm_load(lw, w1h, 1536, 0, 1536);
    __builtin_amdgcn_s_wait_tensorcnt(0);
  }
  __syncthreads();
  const int lane = threadIdx.x & 31, wave = threadIdx.x >> 5;
  const int l15 = lane & 15, half = lane >> 4;
  const int mt = wave & 3, nt = wave >> 2;
  const int ocBase = mt * 16;
  const long p = (long)blockIdx.x * 32 + nt * 16 + l15;
  const int nImg = (int)(p >> 14);
  const int rem = (int)(p & 16383);
  const int oh = rem >> 7, ow = rem & 127;
  const int ihB = oh * 2 - 1, iwB = ow * 2 - 1;
  const float* xb = x + (((long)nImg * 3) << 16);

  auto fb = [&](int k, int n) -> _Float16 {
    (void)n;
    const int ci = k >> 4, r = k & 15;
    const int ih = ihB + (r >> 2), iw = iwB + (r & 3);
    const bool ok = (k < 48) & ((unsigned)ih < 256u) & ((unsigned)iw < 256u);
    const float* pp = ok ? (xb + (((long)ci << 16) + (ih << 8) + iw)) : zeroF;
    return (_Float16)*pp;      // unconditional load; address was selected
  };

  const uint4* arow = (const uint4*)(lw + (ocBase + l15) * 48);   // 6 x uint4
  v8f acc = {0.f, 0.f, 0.f, 0.f, 0.f, 0.f, 0.f, 0.f};
  acc = wmma_f16(acc, fetch_a_row(arow, 0, half), pack_b(lane, 0, fb));
  {
    uint4 z4; z4.x = z4.y = z4.z = z4.w = 0u;
    v8u a1 = cat2(arow[4 + half], z4);                 // k 48..63 zero-padded
    acc = wmma_f16(acc, a1, pack_b(lane, 32, fb));
  }
#pragma unroll
  for (int r = 0; r < 8; ++r) {
    const int oc = ocBase + half * 8 + r;
    float v = acc[r] + b1[oc];
    v = v > 0.f ? v : 0.f;
    z1h[(((long)nImg * 64 + oc) << 14) + (oh << 7) + ow] = (_Float16)v;
  }
}

// ---------------------------------------------------------------------------
// conv2: z1h -> relu -> zeH f16 (32,64,64,64); K=1024 (32 k-steps)
// ---------------------------------------------------------------------------
__global__ __launch_bounds__(256) void k_conv2(const _Float16* __restrict__ z1h,
                                               const _Float16* __restrict__ w2h,
                                               const float* __restrict__ b2,
                                               const _Float16* __restrict__ zeroH,
                                               _Float16* __restrict__ zeH) {
  __shared__ __align__(16) _Float16 lw[32 * 1024];   // 64 KB, 2KB rows
  if (threadIdx.x < 32) {
    tdm_load(lw, w2h + (long)blockIdx.y * 32 * 1024, 16384, 0, 16384);
    __builtin_amdgcn_s_wait_tensorcnt(0);
  }
  __syncthreads();
  const int lane = threadIdx.x & 31, wave = threadIdx.x >> 5;
  const int l15 = lane & 15, half = lane >> 4;
  const int mt = wave & 1, nt = wave >> 1;
  const int ocL = mt * 16;
  const int ocBase = blockIdx.y * 32 + ocL;
  const long p = (long)blockIdx.x * 64 + nt * 16 + l15;
  const int nImg = (int)(p >> 12);
  const int rem = (int)(p & 4095);
  const int oh = rem >> 6, ow = rem & 63;
  const int ihB = oh * 2 - 1, iwB = ow * 2 - 1;
  const _Float16* zb = z1h + (((long)nImg * 64) << 14);

  auto fb = [&](int k, int n) -> _Float16 {
    (void)n;
    const int ci = k >> 4, r = k & 15;
    const int ih = ihB + (r >> 2), iw = iwB + (r & 3);
    const bool ok = ((unsigned)ih < 128u) & ((unsigned)iw < 128u);
    const _Float16* pp = ok ? (zb + (((long)ci << 14) + (ih << 7) + iw)) : zeroH;
    return *pp;
  };

  const uint4* arow = (const uint4*)(lw + (ocL + l15) * 1024);   // 128 x uint4
  v8f acc = {0.f, 0.f, 0.f, 0.f, 0.f, 0.f, 0.f, 0.f};
#pragma unroll 2
  for (int ks = 0; ks < 32; ++ks) {
    acc = wmma_f16(acc, fetch_a_row(arow, ks, half), pack_b(lane, ks * 32, fb));
  }
#pragma unroll
  for (int r = 0; r < 8; ++r) {
    const int oc = ocBase + half * 8 + r;
    float v = acc[r] + b2[oc];
    v = v > 0.f ? v : 0.f;
    zeH[(((long)nImg * 64 + oc) << 12) + (oh << 6) + ow] = (_Float16)v;
  }
}

// ---------------------------------------------------------------------------
// vq: argmin_k ||e_k||^2 - 2 z.e_k over 512 codes for 131072 rows.
// ---------------------------------------------------------------------------
__global__ __launch_bounds__(256) void k_vq(const _Float16* __restrict__ zeH,
                                            const _Float16* __restrict__ cbh,
                                            const float* __restrict__ cbn,
                                            int* __restrict__ idxOut) {
  __shared__ __align__(16) _Float16 lcb[512 * 64];   // 64 KB, 128B rows
  if (threadIdx.x < 32) {
    tdm_load(lcb, cbh, 16384, 0, 16384);
    __builtin_amdgcn_s_wait_tensorcnt(0);
  }
  __syncthreads();
  const int lane = threadIdx.x & 31, wave = threadIdx.x >> 5;
  const int l15 = lane & 15, half = lane >> 4;
  const long rowBase = (long)blockIdx.x * 128 + wave * 16;
  const uint4* arow = (const uint4*)(zeH + (rowBase + l15) * 64);  // 8 x uint4

  v8u a0 = fetch_a_row(arow, 0, half);   // reused across all 32 code tiles
  v8u a1 = fetch_a_row(arow, 1, half);

  float minv[8];
  int mini[8];
#pragma unroll
  for (int r = 0; r < 8; ++r) { minv[r] = 3.4e38f; mini[r] = 0; }

  for (int ct = 0; ct < 32; ++ct) {
    const uint4* brow = (const uint4*)(lcb + (ct * 16 + l15) * 64);
    v8f acc = {0.f, 0.f, 0.f, 0.f, 0.f, 0.f, 0.f, 0.f};
    acc = wmma_f16(acc, a0, fetch_b_row(brow, 0, half));
    acc = wmma_f16(acc, a1, fetch_b_row(brow, 1, half));
    const int code = ct * 16 + l15;
    const float cn = cbn[code];
#pragma unroll
    for (int r = 0; r < 8; ++r) {
      const float s = cn - 2.0f * acc[r];
      if (s < minv[r] || (s == minv[r] && code < mini[r])) {
        minv[r] = s; mini[r] = code;
      }
    }
  }
#pragma unroll
  for (int off = 1; off <= 8; off <<= 1) {
#pragma unroll
    for (int r = 0; r < 8; ++r) {
      const float ov = __shfl_xor(minv[r], off, 32);
      const int   oi = __shfl_xor(mini[r], off, 32);
      if (ov < minv[r] || (ov == minv[r] && oi < mini[r])) {
        minv[r] = ov; mini[r] = oi;
      }
    }
  }
  if (l15 == 0) {
#pragma unroll
    for (int r = 0; r < 8; ++r) idxOut[rowBase + half * 8 + r] = mini[r];
  }
}

// ---------------------------------------------------------------------------
// gather: z_q rows from fp32 codebook -> d_out z_q region + f16 copy
// ---------------------------------------------------------------------------
__global__ __launch_bounds__(256) void k_gather(const int* __restrict__ idx,
                                                const float* __restrict__ cb,
                                                float* __restrict__ zqOut,
                                                _Float16* __restrict__ zqH) {
  const long id = (long)blockIdx.x * 256 + threadIdx.x;   // 2,097,152 threads
  const long m = id >> 4;
  const int t = (int)(id & 15);
  const int code = idx[m];
  const float4 v = ((const float4*)cb)[code * 16 + t];
  ((float4*)zqOut)[m * 16 + t] = v;
  _Float16* zp = zqH + m * 64 + t * 4;
  zp[0] = (_Float16)v.x; zp[1] = (_Float16)v.y;
  zp[2] = (_Float16)v.z; zp[3] = (_Float16)v.w;
}

// ---------------------------------------------------------------------------
// deconv1 (ConvTranspose2d 64->64 s2 p1): parity classes make taps uniform.
// ---------------------------------------------------------------------------
__global__ __launch_bounds__(256) void k_deconv1(const _Float16* __restrict__ zqH,
                                                 const _Float16* __restrict__ dw1h,
                                                 const float* __restrict__ db1,
                                                 const _Float16* __restrict__ zeroH,
                                                 _Float16* __restrict__ hH) {
  __shared__ __align__(16) _Float16 lw[64 * 512];   // 64 KB
  if (threadIdx.x < 32) {
    tdm_load(lw, dw1h + blockIdx.y * 32 * 16, 256, 64, 512);
    __builtin_amdgcn_s_wait_tensorcnt(0);
  }
  __syncthreads();
  const int lane = threadIdx.x & 31, wave = threadIdx.x >> 5;
  const int l15 = lane & 15, half = lane >> 4;
  const int mt = wave & 1, nt = wave >> 1;
  const int ocL = mt * 16;
  const int ocBase = blockIdx.y * 32 + ocL;
  const int py = blockIdx.z >> 1, px = blockIdx.z & 1;
  const int kh0 = 1 - py, kw0 = 1 - px;
  const long s = (long)blockIdx.x * 64 + nt * 16 + l15;
  const int nImg = (int)(s >> 12);
  const int rem = (int)(s & 4095);
  const int yq = rem >> 6, xq = rem & 63;
  const int oy = 2 * yq + py, ox = 2 * xq + px;
  const int iy0 = yq + py, ix0 = xq + px;
  const _Float16* zb = zqH + (((long)nImg * 64) << 12);

  auto fa = [&](int m, int k) -> _Float16 {
    const int ic = k >> 2, t = k & 3;
    const int kh = kh0 + 2 * (t >> 1), kw = kw0 + 2 * (t & 1);
    return lw[ic * 512 + (ocL + m) * 16 + kh * 4 + kw];
  };
  auto fb = [&](int k, int n) -> _Float16 {
    (void)n;
    const int ic = k >> 2, th = (k >> 1) & 1, tw = k & 1;
    const int iy = iy0 - th, ix = ix0 - tw;
    const bool ok = ((unsigned)iy < 64u) & ((unsigned)ix < 64u);
    const _Float16* pp = ok ? (zb + (((long)ic << 12) + (iy << 6) + ix)) : zeroH;
    return *pp;
  };

  v8f acc = {0.f, 0.f, 0.f, 0.f, 0.f, 0.f, 0.f, 0.f};
#pragma unroll
  for (int ks = 0; ks < 8; ++ks) {
    acc = wmma_f16(acc, pack_a(lane, ks * 32, fa), pack_b(lane, ks * 32, fb));
  }
#pragma unroll
  for (int r = 0; r < 8; ++r) {
    const int oc = ocBase + half * 8 + r;
    float v = acc[r] + db1[oc];
    v = v > 0.f ? v : 0.f;
    hH[(((long)nImg * 64 + oc) << 14) + (oy << 7) + ox] = (_Float16)v;
  }
}

// ---------------------------------------------------------------------------
// deconv2 (ConvTranspose2d 64->3 s2 p1) + sigmoid: VALU kernel, branchless
// ---------------------------------------------------------------------------
__global__ __launch_bounds__(256) void k_deconv2(const _Float16* __restrict__ hH,
                                                 const float* __restrict__ dw2,
                                                 const float* __restrict__ db2,
                                                 const _Float16* __restrict__ zeroH,
                                                 float* __restrict__ out) {
  const int id = blockIdx.x * 256 + threadIdx.x;   // 6,291,456 threads
  const int n = id / 196608;
  const int rem = id - n * 196608;
  const int oc = rem >> 16;
  const int pix = rem & 65535;
  const int oy = pix >> 8, ox = pix & 255;
  const int kh0 = (oy + 1) & 1, kw0 = (ox + 1) & 1;
  const int iy0 = (oy + 1 - kh0) >> 1, ix0 = (ox + 1 - kw0) >> 1;
  const int wbase = kh0 * 4 + kw0;
  float acc = db2[oc];
  for (int ic = 0; ic < 64; ++ic) {
    const _Float16* hp = hH + (((long)n * 64 + ic) << 14);
    const float* wp = dw2 + ((ic * 3 + oc) << 4) + wbase;
#pragma unroll
    for (int th = 0; th < 2; ++th) {
      const int iy = iy0 - th;
      const bool vy = (unsigned)iy < 128u;
#pragma unroll
      for (int tw = 0; tw < 2; ++tw) {
        const int ix = ix0 - tw;
        const bool ok = vy & ((unsigned)ix < 128u);
        const _Float16* pp = ok ? (hp + ((iy << 7) + ix)) : zeroH;
        acc += (float)*pp * wp[th * 8 + tw * 2];   // zero-page value kills OOB
      }
    }
  }
  out[id] = 1.0f / (1.0f + __expf(-acc));
}

// ---------------------------------------------------------------------------
extern "C" void kernel_launch(void* const* d_in, const int* in_sizes, int n_in,
                              void* d_out, int out_size, void* d_ws, size_t ws_size,
                              hipStream_t stream) {
  (void)in_sizes; (void)n_in; (void)out_size; (void)ws_size;
  const float* x   = (const float*)d_in[0];
  const float* w1  = (const float*)d_in[1];
  const float* b1  = (const float*)d_in[2];
  const float* w2  = (const float*)d_in[3];
  const float* b2  = (const float*)d_in[4];
  const float* cb  = (const float*)d_in[5];
  const float* dw1 = (const float*)d_in[6];
  const float* db1 = (const float*)d_in[7];
  const float* dw2 = (const float*)d_in[8];
  const float* db2 = (const float*)d_in[9];

  float* xrec  = (float*)d_out;                       // (32,3,256,256)
  float* zqOut = (float*)d_out + 6291456ll;           // (32,64,64,64)

  char* wp = (char*)d_ws;
  size_t off = 0;
  auto take = [&](size_t bytes) -> void* {
    void* r = wp + off;
    off += (bytes + 255) & ~(size_t)255;
    return r;
  };
  _Float16* z1h  = (_Float16*)take(33554432ull * 2);  // (32,64,128,128) f16
  _Float16* zeH  = (_Float16*)take(8388608ull * 2);   // (32,64,64,64)   f16
  _Float16* zqH  = (_Float16*)take(8388608ull * 2);   // (32,64,64,64)   f16
  _Float16* hH   = (_Float16*)take(33554432ull * 2);  // (32,64,128,128) f16
  _Float16* w1h  = (_Float16*)take(3072ull * 2);
  _Float16* w2h  = (_Float16*)take(65536ull * 2);
  _Float16* cbh  = (_Float16*)take(32768ull * 2);
  _Float16* dw1h = (_Float16*)take(65536ull * 2);
  float*    cbn  = (float*)take(512ull * 4);
  int*      vidx = (int*)take(131072ull * 4);
  float*    zpg  = (float*)take(256);                 // zero page (OOB target)

  k_prep   <<<256, 256, 0, stream>>>(w1, w2, cb, dw1, w1h, w2h, cbh, dw1h,
                                     cbn, zpg);
  k_conv1  <<<16384, 256, 0, stream>>>(x, w1h, b1, zpg, z1h);
  k_conv2  <<<dim3(2048, 2), 256, 0, stream>>>(z1h, w2h, b2,
                                               (const _Float16*)zpg, zeH);
  k_vq     <<<1024, 256, 0, stream>>>(zeH, cbh, cbn, vidx);
  k_gather <<<8192, 256, 0, stream>>>(vidx, cb, zqOut, zqH);
  k_deconv1<<<dim3(2048, 2, 4), 256, 0, stream>>>(zqH, dw1h, db1,
                                                  (const _Float16*)zpg, hH);
  k_deconv2<<<24576, 256, 0, stream>>>(hH, dw2, db2, (const _Float16*)zpg, xrec);
}
